// EnhancedDeChunkLayer_63917703299652
// MI455X (gfx1250) — matrix operand order repaired and checked by reading, not compiled, not run on hardware
//
#include <hip/hip_runtime.h>

// Harness contract: kernel_launch only, all launches on `stream`, scratch in d_ws,
// deterministic, no hipMalloc/hipFree/sync-memcpy/events.

#define D_MODEL 1536
#define N_HEADS 12
#define HEAD_D  128
#define BATCH   4
#define SEQ     1024
#define NCHUNK  32
#define ROWS    (BATCH * SEQ)          // 4096
#define FF_DIM  3072

typedef _Float16 half_t;
typedef __attribute__((ext_vector_type(16))) _Float16 v16h;
typedef __attribute__((ext_vector_type(8)))  _Float16 v8h;
typedef __attribute__((ext_vector_type(8)))  float    v8f;

__device__ __forceinline__ v16h join16(v8h lo, v8h hi) {
    return __builtin_shufflevector(lo, hi, 0,1,2,3,4,5,6,7,8,9,10,11,12,13,14,15);
}
#define WMMA_F16(a, b, c) \
    __builtin_amdgcn_wmma_f32_16x16x32_f16(false, (a), false, (b), (short)0, (c), false, false)

__device__ __forceinline__ void wait_ds0() {
#if defined(__has_builtin) && __has_builtin(__builtin_amdgcn_s_wait_dscnt)
    __builtin_amdgcn_s_wait_dscnt(0);
    asm volatile("" ::: "memory");
#else
    asm volatile("s_wait_dscnt 0" ::: "memory");
#endif
}

// ---------------------------------------------------------------- f32 -> f16
__global__ void cvt_f32_f16_kernel(const float* __restrict__ src,
                                   half_t* __restrict__ dst, long n8) {
    long i = (long)blockIdx.x * blockDim.x + threadIdx.x;
    long stride = (long)gridDim.x * blockDim.x;
    for (; i < n8; i += stride) {
        const float4* s = (const float4*)src + 2 * i;
        float4 a = s[0], b = s[1];
        v8h o;
        o[0] = (half_t)a.x; o[1] = (half_t)a.y; o[2] = (half_t)a.z; o[3] = (half_t)a.w;
        o[4] = (half_t)b.x; o[5] = (half_t)b.y; o[6] = (half_t)b.z; o[7] = (half_t)b.w;
        ((v8h*)dst)[i] = o;
    }
}

// ---------------------------------------------------------------- LayerNorm
__global__ __launch_bounds__(256) void ln_kernel(
    const float* __restrict__ x, const float* __restrict__ g,
    const float* __restrict__ b, half_t* __restrict__ outh,
    float* __restrict__ outf, int d) {
    long row = blockIdx.x;
    const float* xr = x + row * d;
    int t = threadIdx.x;
    float s = 0.f, s2 = 0.f;
    for (int i = t; i < d; i += 256) { float v = xr[i]; s += v; s2 += v * v; }
    __shared__ float rs[256], rs2[256];
    rs[t] = s; rs2[t] = s2;
    __syncthreads();
    for (int off = 128; off > 0; off >>= 1) {
        if (t < off) { rs[t] += rs[t + off]; rs2[t] += rs2[t + off]; }
        __syncthreads();
    }
    float mean = rs[0] / d;
    float var  = rs2[0] / d - mean * mean;
    float inv  = rsqrtf(var + 1e-5f);
    for (int i = t; i < d; i += 256) {
        float v = (xr[i] - mean) * inv * g[i] + b[i];
        if (outh) outh[row * d + i] = (half_t)v;
        if (outf) outf[row * d + i] = v;
    }
}

// ---------------------------------------------------------------- ragged expand
__global__ __launch_bounds__(256) void expand_kernel(
    const float* __restrict__ pc, const int* __restrict__ cinfo,
    float* __restrict__ xout, half_t* __restrict__ xh) {
    int bt = blockIdx.x;
    int b = bt >> 10, tpos = bt & 1023;
    const int* c = cinfo + b * NCHUNK;
    int cid = 0, start = 0, run = 0;
#pragma unroll
    for (int i = 0; i < NCHUNK; i++) {
        run += c[i];
        if (run <= tpos) { cid++; start = run; }
    }
    int cs = c[cid];
    float tl = (float)(tpos - start);
    float scale = cs > 1 ? 1.0f + 0.1f * tl / (float)(cs - 1) : 1.0f;
    const float* src = pc + ((long)b * NCHUNK + cid) * D_MODEL;
    long o = ((long)b * SEQ + tpos) * D_MODEL;
    for (int dd = threadIdx.x; dd < D_MODEL; dd += 256) {
        float v = src[dd] * scale;
        xout[o + dd] = v;
        xh[o + dd] = (half_t)v;
    }
}

// ---------------------------------------------------------------- WMMA GEMM
// C[M,N] = epilogue(A[M,K](f16) @ W[N,K]^T(f16) + bias).
// Block tile 128x128, 8 waves (4 along M, 2 along N), wave tile 32x64 =
// 2x4 v_wmma_f32_16x16x32_f16 per K-step. Software-pipelined K loop:
// next tile's global loads are issued before the current compute phase.
__global__ __launch_bounds__(256) void gemm_kernel(
    const half_t* __restrict__ A, const half_t* __restrict__ W,
    const float* __restrict__ bias, const float* __restrict__ resid,
    float* __restrict__ outf, half_t* __restrict__ outh,
    int M, int N, int K, int gelu) {
    __shared__ half_t As[128][40];    // [m][k], padded rows (80B)
    __shared__ half_t Bs[32][136];    // [k][n], padded rows (272B)
    int nb = N >> 7;
    int bm = blockIdx.x / nb, bn = blockIdx.x % nb;
    int m0 = bm << 7, n0 = bn << 7;
    int t = threadIdx.x;
    int w = t >> 5, lane = t & 31, lm = lane & 15, g = lane >> 4;
    int wm = w & 3, wn = w >> 2;

    // staging addresses (fixed per thread)
    int arow = t >> 1, ac16 = (t & 1) << 4;      // A: 128 rows x 32 k
    int brow = t >> 1, bk16 = (t & 1) << 4;      // B: 128 n   x 32 k
    const half_t* aptr = A + (long)(m0 + arow) * K + ac16;
    const half_t* wptr = W + (long)(n0 + brow) * K + bk16;

    v8f acc[2][4] = {};

    // prologue: tile kk=0 -> LDS
    {
        v8h a0 = *(const v8h*)aptr, a1 = *(const v8h*)(aptr + 8);
        v8h b0 = *(const v8h*)wptr, b1 = *(const v8h*)(wptr + 8);
        *(v8h*)&As[arow][ac16]     = a0;
        *(v8h*)&As[arow][ac16 + 8] = a1;
#pragma unroll
        for (int i = 0; i < 8; i++) { Bs[bk16 + i][brow] = b0[i]; Bs[bk16 + 8 + i][brow] = b1[i]; }
    }
    __syncthreads();

    for (int kk = 0; kk < K; kk += 32) {
        bool has_next = (kk + 32) < K;
        v8h na0, na1, nb0, nb1;
        if (has_next) {
            const half_t* ap = aptr + kk + 32;
            const half_t* wp = wptr + kk + 32;
            na0 = *(const v8h*)ap; na1 = *(const v8h*)(ap + 8);
            nb0 = *(const v8h*)wp; nb1 = *(const v8h*)(wp + 8);
            if (kk + 64 < K) {  // prefetch K+64 tile into cache (global_prefetch_b8)
                __builtin_prefetch(aptr + kk + 64, 0, 3);
                __builtin_prefetch(wptr + kk + 64, 0, 3);
            }
        }

        // compute current tile from LDS
        v16h af[2], bf[4];
#pragma unroll
        for (int i = 0; i < 2; i++) {
            int m = wm * 32 + i * 16 + lm;
            af[i] = join16(*(const v8h*)&As[m][8 * g],
                           *(const v8h*)&As[m][16 + 8 * g]);
        }
#pragma unroll
        for (int j = 0; j < 4; j++) {
            int n = wn * 64 + j * 16 + lm;
#pragma unroll
            for (int s = 0; s < 16; s++) bf[j][s] = Bs[(s & 7) + 8 * g + ((s >> 3) << 4)][n];
        }
#pragma unroll
        for (int i = 0; i < 2; i++)
#pragma unroll
            for (int j = 0; j < 4; j++)
                acc[i][j] = WMMA_F16(af[i], bf[j], acc[i][j]);
        __syncthreads();

        if (has_next) {
            *(v8h*)&As[arow][ac16]     = na0;
            *(v8h*)&As[arow][ac16 + 8] = na1;
#pragma unroll
            for (int i = 0; i < 8; i++) { Bs[bk16 + i][brow] = nb0[i]; Bs[bk16 + 8 + i][brow] = nb1[i]; }
            __syncthreads();
        }
    }

#pragma unroll
    for (int i = 0; i < 2; i++)
#pragma unroll
        for (int j = 0; j < 4; j++) {
            int n = n0 + wn * 64 + j * 16 + lm;
            float bv = bias ? bias[n] : 0.f;
#pragma unroll
            for (int r = 0; r < 8; r++) {
                int m = m0 + wm * 32 + i * 16 + r + 8 * g;
                float v = acc[i][j][r] + bv;
                if (gelu) v = 0.5f * v * (1.f + erff(v * 0.70710678118654752f));
                if (resid) v += resid[(long)m * N + n];
                if (outf) outf[(long)m * N + n] = v;
                if (outh) outh[(long)m * N + n] = (half_t)v;
            }
        }
}

// ---------------------------------------------------------------- flash attention
// 4 waves / block, each wave owns 16 query rows; software-pipelined loop over
// 32-key chunks with online softmax; QK^T and PV via v_wmma_f32_16x16x32_f16.
__global__ __launch_bounds__(128) void attn_kernel(
    const half_t* __restrict__ Q, const half_t* __restrict__ Kb,
    const half_t* __restrict__ Vb, half_t* __restrict__ O) {
    __shared__ half_t KT[128][40];       // [d][key]
    __shared__ half_t Vs[32][136];       // [key][d]
    __shared__ half_t Pst[4][16][40];    // per-wave P staging [m][key]

    int blk = blockIdx.x;
    int qt = blk & 15;
    int h  = (blk >> 4) % N_HEADS;
    int b  = blk / (16 * N_HEADS);
    int t = threadIdx.x;
    int w = t >> 5, lane = t & 31, lm = lane & 15, g = lane >> 4;
    int qbase = qt * 64 + w * 16;

    const half_t* qp = Q + ((long)b * SEQ + qbase) * D_MODEL + h * HEAD_D;
    v16h qf[4];
#pragma unroll
    for (int c = 0; c < 4; c++) {
        const half_t* p = qp + (long)lm * D_MODEL + c * 32 + 8 * g;
        qf[c] = join16(*(const v8h*)p, *(const v8h*)(p + 16));
    }

    // fixed per-thread staging addressing: 4 threads per key, 32 d each
    int key = t >> 2, db = (t & 3) << 5;
    const half_t* kbase = Kb + ((long)b * SEQ + key) * D_MODEL + h * HEAD_D + db;
    const half_t* vbase = Vb + ((long)b * SEQ + key) * D_MODEL + h * HEAD_D + db;

    v8f oacc[8] = {};
    float mrun[8], lsum[8];
#pragma unroll
    for (int r = 0; r < 8; r++) { mrun[r] = -1e30f; lsum[r] = 0.f; }
    const float scale = 0.08838834764831845f;  // 1/sqrt(128)

    // prologue: chunk 0 -> LDS
    {
#pragma unroll
        for (int q8 = 0; q8 < 4; q8++) {
            v8h kv = *(const v8h*)(kbase + q8 * 8);
#pragma unroll
            for (int i = 0; i < 8; i++) KT[db + q8 * 8 + i][key] = kv[i];
            *(v8h*)&Vs[key][db + q8 * 8] = *(const v8h*)(vbase + q8 * 8);
        }
    }
    __syncthreads();

    for (int j = 0; j < 32; j++) {
        bool has_next = j < 31;
        v8h nk[4], nv[4];
        if (has_next) {
            const half_t* kp = kbase + (long)(j + 1) * 32 * D_MODEL;
            const half_t* vp = vbase + (long)(j + 1) * 32 * D_MODEL;
#pragma unroll
            for (int q8 = 0; q8 < 4; q8++) {
                nk[q8] = *(const v8h*)(kp + q8 * 8);
                nv[q8] = *(const v8h*)(vp + q8 * 8);
            }
        }

        // scores S(16x32) = Q(16x128) . K^T, two 16-key subtiles
        v8f sc[2] = {};
#pragma unroll
        for (int nt = 0; nt < 2; nt++)
#pragma unroll
            for (int c = 0; c < 4; c++) {
                v16h bf;
#pragma unroll
                for (int s = 0; s < 16; s++) bf[s] = KT[c * 32 + (s & 7) + 8 * g + ((s >> 3) << 4)][nt * 16 + lm];
                sc[nt] = WMMA_F16(qf[c], bf, sc[nt]);
            }

        // scale + row max across the 16-lane group
        float rmax[8];
#pragma unroll
        for (int r = 0; r < 8; r++) {
            float a = sc[0][r] * scale, c2v = sc[1][r] * scale;
            sc[0][r] = a; sc[1][r] = c2v;
            rmax[r] = fmaxf(a, c2v);
        }
#pragma unroll
        for (int r = 0; r < 8; r++) {
            float v = rmax[r];
            v = fmaxf(v, __shfl_xor(v, 1));
            v = fmaxf(v, __shfl_xor(v, 2));
            v = fmaxf(v, __shfl_xor(v, 4));
            v = fmaxf(v, __shfl_xor(v, 8));
            rmax[r] = v;
        }
        float corr[8];
#pragma unroll
        for (int r = 0; r < 8; r++) {
            float mn = fmaxf(mrun[r], rmax[r]);
            corr[r] = __expf(mrun[r] - mn);
            mrun[r] = mn;
            lsum[r] *= corr[r];
        }
#pragma unroll
        for (int c = 0; c < 8; c++)
#pragma unroll
            for (int r = 0; r < 8; r++) oacc[c][r] *= corr[r];

        float psum[8];
#pragma unroll
        for (int r = 0; r < 8; r++) {
            float p0 = __expf(sc[0][r] - mrun[r]);
            float p1 = __expf(sc[1][r] - mrun[r]);
            sc[0][r] = p0; sc[1][r] = p1;
            psum[r] = p0 + p1;
        }
#pragma unroll
        for (int r = 0; r < 8; r++) {
            float v = psum[r];
            v += __shfl_xor(v, 1);
            v += __shfl_xor(v, 2);
            v += __shfl_xor(v, 4);
            v += __shfl_xor(v, 8);
            lsum[r] += v;
        }

        // P: C-layout -> A-layout via per-wave LDS stage (same-wave DS ordering)
#pragma unroll
        for (int nt = 0; nt < 2; nt++)
#pragma unroll
            for (int r = 0; r < 8; r++)
                Pst[w][r + 8 * g][nt * 16 + lm] = (half_t)sc[nt][r];
        wait_ds0();
        v16h pf = join16(*(const v8h*)&Pst[w][lm][8 * g],
                         *(const v8h*)&Pst[w][lm][16 + 8 * g]);

        // O += P(16x32) . V(32x128)
#pragma unroll
        for (int c = 0; c < 8; c++) {
            v16h vf;
#pragma unroll
            for (int s = 0; s < 16; s++) vf[s] = Vs[(s & 7) + 8 * g + ((s >> 3) << 4)][c * 16 + lm];
            oacc[c] = WMMA_F16(pf, vf, oacc[c]);
        }
        __syncthreads();

        if (has_next) {
#pragma unroll
            for (int q8 = 0; q8 < 4; q8++) {
#pragma unroll
                for (int i = 0; i < 8; i++) KT[db + q8 * 8 + i][key] = nk[q8][i];
                *(v8h*)&Vs[key][db + q8 * 8] = nv[q8];
            }
            __syncthreads();
        }
    }

    half_t* op = O + ((long)b * SEQ + qbase) * D_MODEL + h * HEAD_D;
#pragma unroll
    for (int r = 0; r < 8; r++) lsum[r] = 1.0f / lsum[r];
#pragma unroll
    for (int c = 0; c < 8; c++)
#pragma unroll
        for (int r = 0; r < 8; r++) {
            int m = r + 8 * g;
            int d = c * 16 + lm;
            op[(long)m * D_MODEL + d] = (half_t)(oacc[c][r] * lsum[r]);
        }
}

// ---------------------------------------------------------------- host orchestration
extern "C" void kernel_launch(void* const* d_in, const int* in_sizes, int n_in,
                              void* d_out, int out_size, void* d_ws, size_t ws_size,
                              hipStream_t stream) {
    (void)in_sizes; (void)n_in; (void)out_size; (void)ws_size;
    const int D = D_MODEL, FF = FF_DIM;

    char* wsp = (char*)d_ws;
    size_t off = 0;
    auto alloc = [&](size_t bytes) -> void* {
        void* p = wsp + off;
        off += (bytes + 255) & ~(size_t)255;
        return p;
    };

    float*  xf32  = (float*)alloc((size_t)ROWS * D * 4);
    half_t* xin   = (half_t*)alloc((size_t)ROWS * D * 2);   // layer-input f16 (cross-attn memory)
    half_t* hbuf  = (half_t*)alloc((size_t)ROWS * D * 2);   // LN outputs
    half_t* qb    = (half_t*)alloc((size_t)ROWS * D * 2);
    half_t* kb    = (half_t*)alloc((size_t)ROWS * D * 2);
    half_t* vb    = (half_t*)alloc((size_t)ROWS * D * 2);
    half_t* ob    = (half_t*)alloc((size_t)ROWS * D * 2);
    half_t* tbuf  = (half_t*)alloc((size_t)ROWS * FF * 2);
    float*  pcraw = (float*)alloc((size_t)BATCH * NCHUNK * D * 4);
    half_t* w8[8];
    for (int i = 0; i < 8; i++) w8[i] = (half_t*)alloc((size_t)D * D * 2);
    half_t* wf1 = (half_t*)alloc((size_t)FF * D * 2);
    half_t* wf2 = (half_t*)alloc((size_t)D * FF * 2);

    auto cvt = [&](const void* src, half_t* dst, long n) {
        long n8 = n / 8;
        int grid = (int)((n8 + 255) / 256);
        if (grid > 4096) grid = 4096;
        cvt_f32_f16_kernel<<<grid, 256, 0, stream>>>((const float*)src, dst, n8);
    };
    auto gemm = [&](const half_t* A, const half_t* W, const void* bias,
                    const float* res, float* of, half_t* oh,
                    int M, int N, int K, int gelu) {
        dim3 grid((M / 128) * (N / 128));
        gemm_kernel<<<grid, 256, 0, stream>>>(A, W, (const float*)bias, res, of, oh,
                                              M, N, K, gelu);
    };
    auto ln = [&](const float* x, const void* g, const void* b,
                  half_t* oh, float* of, int rows) {
        ln_kernel<<<rows, 256, 0, stream>>>(x, (const float*)g, (const float*)b, oh, of, D);
    };
    auto IN = [&](int i) -> const void* { return d_in[i]; };

    // ---- reconstruction net: GELU(hr@w1^T+b1)@w2^T+b2 -> LN -> expand
    cvt(IN(2), wf1, (long)FF * D);               // rec.w1
    cvt(IN(4), wf2, (long)D * FF);               // rec.w2
    cvt(IN(0), hbuf, (long)BATCH * NCHUNK * D);  // hierarchical_repr -> f16
    gemm(hbuf, wf1, IN(3), nullptr, nullptr, tbuf, BATCH * NCHUNK, FF, D, 1);
    gemm(tbuf, wf2, IN(5), nullptr, pcraw, nullptr, BATCH * NCHUNK, D, FF, 0);
    ln_kernel<<<BATCH * NCHUNK, 256, 0, stream>>>(pcraw, (const float*)IN(6),
                                                  (const float*)IN(7), nullptr, pcraw, D);
    expand_kernel<<<ROWS, 256, 0, stream>>>(pcraw, (const int*)IN(1), xf32, xin);

    // ---- decoder layers
    for (int l = 0; l < 4; l++) {
        int B0 = 8 + 26 * l;
        // weights -> f16
        cvt(IN(B0 + 0),  w8[0], (long)D * D);   // sa.wq
        cvt(IN(B0 + 2),  w8[1], (long)D * D);   // sa.wk
        cvt(IN(B0 + 4),  w8[2], (long)D * D);   // sa.wv
        cvt(IN(B0 + 6),  w8[3], (long)D * D);   // sa.wo
        cvt(IN(B0 + 8),  w8[4], (long)D * D);   // ca.wq
        cvt(IN(B0 + 10), w8[5], (long)D * D);   // ca.wk
        cvt(IN(B0 + 12), w8[6], (long)D * D);   // ca.wv
        cvt(IN(B0 + 14), w8[7], (long)D * D);   // ca.wo
        cvt(IN(B0 + 22), wf1, (long)FF * D);    // ffn.w1
        cvt(IN(B0 + 24), wf2, (long)D * FF);    // ffn.w2

        // self-attention (norm_first)
        ln(xf32, IN(B0 + 16), IN(B0 + 17), hbuf, nullptr, ROWS);
        gemm(hbuf, w8[0], IN(B0 + 1), nullptr, nullptr, qb, ROWS, D, D, 0);
        gemm(hbuf, w8[1], IN(B0 + 3), nullptr, nullptr, kb, ROWS, D, D, 0);
        gemm(hbuf, w8[2], IN(B0 + 5), nullptr, nullptr, vb, ROWS, D, D, 0);
        attn_kernel<<<BATCH * N_HEADS * (SEQ / 64), 128, 0, stream>>>(qb, kb, vb, ob);
        gemm(ob, w8[3], IN(B0 + 7), xf32, xf32, nullptr, ROWS, D, D, 0);

        // cross-attention over layer input (xin)
        ln(xf32, IN(B0 + 18), IN(B0 + 19), hbuf, nullptr, ROWS);
        gemm(hbuf, w8[4], IN(B0 + 9),  nullptr, nullptr, qb, ROWS, D, D, 0);
        gemm(xin,  w8[5], IN(B0 + 11), nullptr, nullptr, kb, ROWS, D, D, 0);
        gemm(xin,  w8[6], IN(B0 + 13), nullptr, nullptr, vb, ROWS, D, D, 0);
        attn_kernel<<<BATCH * N_HEADS * (SEQ / 64), 128, 0, stream>>>(qb, kb, vb, ob);
        gemm(ob, w8[7], IN(B0 + 15), xf32, xf32, nullptr, ROWS, D, D, 0);

        // FFN
        ln(xf32, IN(B0 + 20), IN(B0 + 21), hbuf, nullptr, ROWS);
        gemm(hbuf, wf1, IN(B0 + 23), nullptr, nullptr, tbuf, ROWS, FF, D, 1);
        float*  fo   = (l == 3) ? (float*)d_out : xf32;
        half_t* fo16 = (l == 3) ? nullptr : xin;   // becomes next layer's memory
        gemm(tbuf, wf2, IN(B0 + 25), xf32, fo, fo16, ROWS, D, FF, 0);
    }
}
